// LSTM_63376537420327
// MI455X (gfx1250) — matrix operand order repaired and checked by reading, not compiled, not run on hardware
//
#include <hip/hip_runtime.h>
#include <math.h>

typedef __attribute__((ext_vector_type(16))) __bf16 v16bf;
typedef __attribute__((ext_vector_type(8)))  __bf16 v8bf;
typedef __attribute__((ext_vector_type(8)))  float  v8f;

static constexpr int kB = 256;   // batch
static constexpr int kS = 512;   // sequence length
static constexpr int kH = 1024;  // hidden
static constexpr int kC = 10;    // classes

static constexpr int kNSlice  = 16;             // h-columns per persistent block
static constexpr int kNBlocks = kH / kNSlice;   // 64 persistent blocks
static constexpr int kThreads = 512;            // 16 waves -> all 256 batch rows
static constexpr int kRowPad  = kH + 8;         // LDS row stride (bf16): bank-conflict pad
static constexpr size_t kLdsBytes = (size_t)4 * kNSlice * kRowPad * sizeof(__bf16); // ~129 KB

union ABreg { v16bf v; v8bf h[2]; };

__device__ __forceinline__ __bf16 f2bf(float f) {
    union { float f; unsigned u; } v; v.f = f;
    unsigned r = (v.u + 0x7FFFu + ((v.u >> 16) & 1u)) >> 16;
    union { unsigned short s; __bf16 b; } o; o.s = (unsigned short)r;
    return o.b;
}

// CDNA5 has a native V_TANH_F32 transcendental (TRANS32 op, co-executes with
// the matrix pipe). Use it for tanh AND for sigmoid via the identity
// sigmoid(x) = 0.5*tanh(x/2) + 0.5, killing the OCML exp/divergence ladders.
__device__ __forceinline__ float fast_tanh(float x) {
#if __has_builtin(__builtin_amdgcn_tanhf)
    return __builtin_amdgcn_tanhf(x);
#elif __has_builtin(__builtin_amdgcn_tanh_f32)
    return __builtin_amdgcn_tanh_f32(x);
#else
    return tanhf(x);
#endif
}

__device__ __forceinline__ float fast_sigmoid(float x) {
#if __has_builtin(__builtin_amdgcn_tanhf) || __has_builtin(__builtin_amdgcn_tanh_f32)
    return __builtin_fmaf(0.5f, fast_tanh(0.5f * x), 0.5f);
#else
    return 1.0f / (1.0f + __expf(-x));
#endif
}

// ---------------------------------------------------------------------------
// zero h0 (bf16), c0 (fp32), and the per-step barrier counters
// ---------------------------------------------------------------------------
__global__ void init_state_kernel(float* __restrict__ c_st,
                                  __bf16* __restrict__ h_bf,
                                  unsigned* __restrict__ cnt, int n) {
    int i = blockIdx.x * blockDim.x + threadIdx.x;
    if (i < n) {
        c_st[i] = 0.0f;
        union { unsigned short s; __bf16 b; } z; z.s = 0;
        h_bf[i] = z.b;
    }
    if (i < kS) cnt[i] = 0u;
}

// ---------------------------------------------------------------------------
// Persistent LSTM: one launch runs all 512 timesteps.
//   - Weight slice [4 gates][16 cols][1024 k] pinned in LDS (bf16, padded rows)
//   - A (h_{t-1}) from global bf16 ping-pong (L2-resident)
//   - grid-wide release/acquire barrier per step via cnt[t]
// Block: 512 threads = 16 waves; wave w owns batch rows [16w, 16w+16).
// Grid: 64 blocks; block bx owns h-columns [16bx, 16bx+16) for all 4 gates.
// ---------------------------------------------------------------------------
__global__ __launch_bounds__(kThreads)
void lstm_persist_kernel(const float* __restrict__ wgh, const float* __restrict__ wih,
                         const float* __restrict__ wfh, const float* __restrict__ woh,
                         __bf16* __restrict__ hb0, __bf16* __restrict__ hb1,
                         float* __restrict__ h_f32, float* __restrict__ c_st,
                         const float* __restrict__ x,     // [kB][kS]
                         const float* __restrict__ wgx, const float* __restrict__ wix,
                         const float* __restrict__ wfx, const float* __restrict__ wox,
                         const float* __restrict__ bg,  const float* __restrict__ bi,
                         const float* __restrict__ bf_, const float* __restrict__ bo,
                         unsigned* __restrict__ cnt)
{
    extern __shared__ __align__(16) char smem_raw[];
    __bf16* Wlds = (__bf16*)smem_raw;   // [4*kNSlice][kRowPad]

    const int nbase = blockIdx.x * kNSlice;

    // ---- stage weight slice into LDS, converting fp32 -> bf16 (RNE) ----
    {
        const float* wsrc[4] = { wgh, wih, wfh, woh };
        // 64 rows (g*16+nloc) x 128 chunks of 8 elements
        for (int idx = threadIdx.x; idx < 4 * kNSlice * (kH / 8); idx += kThreads) {
            const int chunk = idx & (kH / 8 - 1);   // 0..127
            const int row   = idx >> 7;             // 0..63
            const int g     = row >> 4;
            const int nloc  = row & (kNSlice - 1);
            const float* src = wsrc[g] + (size_t)(nbase + nloc) * kH + (chunk << 3);
            v8bf o;
            #pragma unroll
            for (int j = 0; j < 8; ++j) o[j] = f2bf(src[j]);
            *(v8bf*)(Wlds + (size_t)row * kRowPad + (chunk << 3)) = o;
        }
    }
    __syncthreads();

    const int lane = threadIdx.x & 31;
    const int wave = threadIdx.x >> 5;   // 0..15
    const int hi   = lane >> 4;          // lane half (0/1)
    const int l16  = lane & 15;
    const int m0   = wave * 16;          // batch-row base of this wave
    const int n    = nbase + l16;        // this lane's h-column (C/D layout: N = lane%16)

    // per-column constants live in VGPRs for the whole sequence
    const float wg = wgx[n], wi = wix[n], wf = wfx[n], wo = wox[n];
    const float cg = bg[n],  ci = bi[n],  cf = bf_[n], co = bo[n];

    // A layout (16-bit A 16x32, ISA 7.12.2): lane row = l16,
    // K runs [koffA, koffA+8) and [koffA+16, koffA+24), koffA = hi*8.
    const int koffA = hi * 8;

    for (int t = 0; t < kS; ++t) {
        const __bf16* hs = (t & 1) ? hb1 : hb0;
        __bf16*       hd = (t & 1) ? hb0 : hb1;

        v8f acc[4];
        { v8f z = {}; acc[0] = z; acc[1] = z; acc[2] = z; acc[3] = z; }

        const __bf16* aRow = hs + (size_t)(m0 + l16) * kH + koffA;

        for (int k0 = 0; k0 < kH; k0 += 32) {
            ABreg a;
            a.h[0] = *(const v8bf*)(aRow + k0);
            a.h[1] = *(const v8bf*)(aRow + k0 + 16);
            #pragma unroll
            for (int g = 0; g < 4; ++g) {
                // B layout (32x16): lane col = l16; K runs [hi*8, hi*8+8)
                // and [hi*8+16, hi*8+24) -> 2 ds_load_b128, matching A.
                const __bf16* bp = Wlds + (size_t)(g * kNSlice + l16) * kRowPad
                                        + k0 + hi * 8;
                ABreg b;
                b.h[0] = *(const v8bf*)bp;
                b.h[1] = *(const v8bf*)(bp + 16);
                acc[g] = __builtin_amdgcn_wmma_f32_16x16x32_bf16(
                    false, a.v, false, b.v, (short)0, acc[g], false, false);
            }
        }

        // Elementwise gate math. C/D layout: VGPR r, lanes 0-15 -> (M=r, N=lane),
        // lanes 16-31 -> (M=r+8, N=lane-16).
        #pragma unroll
        for (int r = 0; r < 8; ++r) {
            const int b  = m0 + r + 8 * hi;
            const float xv = x[(size_t)b * kS + t];
            const float pg = acc[0][r] + xv * wg + cg;
            const float pi = acc[1][r] + xv * wi + ci;
            const float pf = acc[2][r] + xv * wf + cf;
            const float po = acc[3][r] + xv * wo + co;
            const float gv = fast_tanh(pg);
            const float iv = fast_sigmoid(pi);
            const float fv = fast_sigmoid(pf);
            const float ov = fast_sigmoid(po);
            const size_t idx = (size_t)b * kH + n;
            const float cnew = gv * iv + c_st[idx] * fv;
            const float hnew = fast_tanh(cnew) * ov;
            c_st[idx] = cnew;
            hd[idx]   = (__bf16)hnew;
            if (t == kS - 1) h_f32[idx] = hnew;
        }

        // ---- grid-wide barrier: release writes, wait for all 64 blocks ----
        __threadfence();
        __syncthreads();
        if (threadIdx.x == 0) {
            __hip_atomic_fetch_add(&cnt[t], 1u, __ATOMIC_RELEASE,
                                   __HIP_MEMORY_SCOPE_AGENT);
            while (__hip_atomic_load(&cnt[t], __ATOMIC_ACQUIRE,
                                     __HIP_MEMORY_SCOPE_AGENT) < (unsigned)gridDim.x)
                __builtin_amdgcn_s_sleep(1);
        }
        __syncthreads();
        __threadfence();   // drop stale L0 lines in every wave before next step's A reads
    }
}

// ---------------------------------------------------------------------------
// Final projection: out[b,c] = sum_h hT[b,h] * w_ph[c,h] + b_p[c]
// ---------------------------------------------------------------------------
__global__ void proj_kernel(const float* __restrict__ hT,
                            const float* __restrict__ wph,
                            const float* __restrict__ bp,
                            float* __restrict__ out)
{
    int idx = blockIdx.x * blockDim.x + threadIdx.x;
    if (idx >= kB * kC) return;
    int b = idx / kC;
    int c = idx % kC;
    const float* hv = hT  + (size_t)b * kH;
    const float* wv = wph + (size_t)c * kH;
    float s = 0.0f;
    for (int h = 0; h < kH; ++h) s += hv[h] * wv[h];
    out[idx] = s + bp[c];
}

// ---------------------------------------------------------------------------
extern "C" void kernel_launch(void* const* d_in, const int* in_sizes, int n_in,
                              void* d_out, int out_size, void* d_ws, size_t ws_size,
                              hipStream_t stream) {
    (void)in_sizes; (void)n_in; (void)out_size; (void)ws_size;

    const float* x   = (const float*)d_in[0];
    const float* wgx = (const float*)d_in[1];
    const float* wix = (const float*)d_in[2];
    const float* wfx = (const float*)d_in[3];
    const float* wox = (const float*)d_in[4];
    const float* wgh = (const float*)d_in[5];
    const float* wih = (const float*)d_in[6];
    const float* wfh = (const float*)d_in[7];
    const float* woh = (const float*)d_in[8];
    const float* wph = (const float*)d_in[9];
    const float* bg  = (const float*)d_in[10];
    const float* bi  = (const float*)d_in[11];
    const float* bfv = (const float*)d_in[12];
    const float* bo  = (const float*)d_in[13];
    const float* bp  = (const float*)d_in[14];
    float* out = (float*)d_out;

    // Workspace layout (~3 MB)
    char* ws = (char*)d_ws;
    size_t off = 0;
    __bf16* hb0 = (__bf16*)(ws + off); off += (size_t)kB * kH * sizeof(__bf16);
    __bf16* hb1 = (__bf16*)(ws + off); off += (size_t)kB * kH * sizeof(__bf16);
    float*  hf  = (float*)(ws + off);  off += (size_t)kB * kH * sizeof(float);
    float*  cs  = (float*)(ws + off);  off += (size_t)kB * kH * sizeof(float);
    unsigned* cnt = (unsigned*)(ws + off);

    // allow >64KB dynamic LDS for the persistent kernel (no-op if not needed)
    (void)hipFuncSetAttribute((const void*)lstm_persist_kernel,
                              hipFuncAttributeMaxDynamicSharedMemorySize,
                              (int)kLdsBytes);

    const int thr = 256;
    const int nS = kB * kH;
    init_state_kernel<<<(nS + thr - 1) / thr, thr, 0, stream>>>(cs, hb0, cnt, nS);

    lstm_persist_kernel<<<dim3(kNBlocks), kThreads, kLdsBytes, stream>>>(
        wgh, wih, wfh, woh, hb0, hb1, hf, cs,
        x, wgx, wix, wfx, wox, bg, bi, bfv, bo, cnt);

    proj_kernel<<<(kB * kC + thr - 1) / thr, thr, 0, stream>>>(hf, wph, bp, out);
}
